// FusedExperts_50208167690348
// MI455X (gfx1250) — compile-verified, matrix operand order checked by reading, not run on hardware
//
#include <hip/hip_runtime.h>

// ---------------------------------------------------------------------------
// Problem constants: B=4, S=2048, H=1024, F=2048, E=8, K=2
// ---------------------------------------------------------------------------
#define NTOK   8192          // B*S tokens
#define HDIM   1024          // hidden size H
#define FDIM   2048          // ffn size F
#define NEXP   8             // experts
#define TOPK   2
#define NPAIR  (NTOK * TOPK) // 16384 (token, slot) pairs

#define TM     128           // token tile (rows)
#define TN     64            // output-column tile
#define KB     32            // WMMA K step (bf16 16x16x32)
#define MAXTT  (NPAIR / TM)  // 128 worst-case token tiles per expert

typedef __bf16 bf16;
typedef __attribute__((ext_vector_type(4)))  float v4f;
typedef __attribute__((ext_vector_type(8)))  float v8f;
typedef __attribute__((ext_vector_type(4)))  bf16  v4bf;
typedef __attribute__((ext_vector_type(8)))  bf16  v8bf;
typedef __attribute__((ext_vector_type(16))) bf16  v16bf;

// ---------------------------------------------------------------------------
// CDNA5 async copy: 16B global -> LDS, tracked by ASYNCcnt.
// ---------------------------------------------------------------------------
__device__ __forceinline__ void async_copy16(const bf16* lds_dst, const bf16* gsrc) {
  unsigned           l = (unsigned)(size_t)lds_dst;
  unsigned long long g = (unsigned long long)(size_t)gsrc;
  asm volatile("global_load_async_to_lds_b128 %0, %1, off"
               :: "v"(l), "v"(g) : "memory");
}

#if __has_builtin(__builtin_amdgcn_s_wait_asynccnt)
#define WAIT_ASYNC(n) __builtin_amdgcn_s_wait_asynccnt(n)
#else
#define WAIT_ASYNC(n) asm volatile("s_wait_asynccnt %0" :: "n"(n) : "memory")
#endif

#if __has_builtin(__builtin_amdgcn_s_wait_tensorcnt)
#define WAIT_TENSOR(n) __builtin_amdgcn_s_wait_tensorcnt(n)
#else
#define WAIT_TENSOR(n) asm volatile("s_wait_tensorcnt %0" :: "n"(n) : "memory")
#endif

// ---------------------------------------------------------------------------
// Tensor Data Mover: one DMA for a [tileRows x 32] bf16 tile from a row-major
// matrix (row stride = strideElems) into contiguous LDS. Rows >= tensorRows
// are zero-filled by TDM OOB handling (ISA 8.2). Issued per-wave (EXEC
// ignored), tracked by TENSORcnt. D# layout per ISA 8.3-8.6:
//   g0: count=1 | lds_addr | global_addr[56:0] | type=2
//   g1: data_size=2B | tensor_dim0=strideElems | tensor_dim1=tensorRows |
//       tile_dim0=32 | tile_dim1=tileRows | tensor_dim0_stride=strideElems
// This toolchain's builtin takes 6 args (g0, g1, g2, g3, g4, cpol).
// ---------------------------------------------------------------------------
#if __has_builtin(__builtin_amdgcn_tensor_load_to_lds)
#define HAVE_TDM 1
typedef __attribute__((ext_vector_type(4))) unsigned tdm_g0;
typedef __attribute__((ext_vector_type(8))) int      tdm_g1;
typedef __attribute__((ext_vector_type(4))) int      tdm_g2;
typedef __attribute__((ext_vector_type(8))) int      tdm_g4;

__device__ __forceinline__ void tdm_load_tile(const bf16* gsrc, const bf16* lds_dst,
                                              unsigned strideElems, unsigned tensorRows,
                                              unsigned tileRows) {
  unsigned long long ga = (unsigned long long)(size_t)gsrc;
  unsigned           la = (unsigned)(size_t)lds_dst;
  tdm_g0 g0;
  g0[0] = 1u;                                          // count=1 (valid user D#)
  g0[1] = la;                                          // lds_addr
  g0[2] = (unsigned)ga;                                // global_addr[31:0]
  g0[3] = ((unsigned)(ga >> 32) & 0x01FFFFFFu) | 0x80000000u; // ga[56:32], type=2
  tdm_g1 g1;
  g1[0] = 0x00010000;                                  // data_size=1 (2 bytes)
  g1[1] = (int)((strideElems & 0xFFFFu) << 16);        // tensor_dim0[15:0]
  g1[2] = (int)(((strideElems >> 16) & 0xFFFFu)        // tensor_dim0[31:16]
        | ((tensorRows & 0xFFFFu) << 16));             // tensor_dim1[15:0]
  g1[3] = (int)(((tensorRows >> 16) & 0xFFFFu)         // tensor_dim1[31:16]
        | ((unsigned)KB << 16));                       // tile_dim0=32
  g1[4] = (int)(tileRows & 0xFFFFu);                   // tile_dim1, tile_dim2=0
  g1[5] = (int)strideElems;                            // tensor_dim0_stride[31:0]
  g1[6] = 0;                                           // stride hi, dim1_stride lo
  g1[7] = 0;
  tdm_g2 z4 = {0, 0, 0, 0};
  tdm_g4 z8 = {0, 0, 0, 0, 0, 0, 0, 0};
  __builtin_amdgcn_tensor_load_to_lds(g0, g1, z4, z4, z8, 0);
}
#else
#define HAVE_TDM 0
#endif

// Load one 16x32 bf16 WMMA fragment (two 16B LDS reads per ISA 7.12.2).
__device__ __forceinline__ v16bf load_frag(const bf16* __restrict__ row, int kh) {
  v8bf lo = *(const v8bf*)(row + kh);
  v8bf hi = *(const v8bf*)(row + 16 + kh);
  return __builtin_shufflevector(lo, hi, 0, 1, 2, 3, 4, 5, 6, 7,
                                 8, 9, 10, 11, 12, 13, 14, 15);
}

#define WMMA_BF16(ACC, A, B) \
  __builtin_amdgcn_wmma_f32_16x16x32_bf16(false, (A), false, (B), (short)0, (ACC), false, false)

// ---------------------------------------------------------------------------
// Routing: count / scan / scatter
// ---------------------------------------------------------------------------
__global__ void zero_meta_k(int* counts, int* cursors) {
  int t = threadIdx.x;
  if (t < NEXP) { counts[t] = 0; cursors[t] = 0; }
}

__global__ void count_k(const int* __restrict__ sel, int* __restrict__ counts) {
  int p = blockIdx.x * blockDim.x + threadIdx.x;
  if (p < NPAIR) atomicAdd(&counts[sel[p]], 1);
}

__global__ void scan_k(const int* __restrict__ counts, int* __restrict__ offsets) {
  if (threadIdx.x == 0) {
    int acc = 0;
    for (int e = 0; e < NEXP; ++e) { offsets[e] = acc; acc += counts[e]; }
    offsets[NEXP] = acc;
  }
}

__global__ void scatter_k(const int* __restrict__ sel,
                          const int* __restrict__ offsets,
                          int* cursors, int* __restrict__ bucket) {
  int p = blockIdx.x * blockDim.x + threadIdx.x;
  if (p < NPAIR) {
    int e = sel[p];
    int pos = atomicAdd(&cursors[e], 1);
    bucket[offsets[e] + pos] = p;
  }
}

// fp32 -> bf16, 4 elements/thread
__global__ void cvt_f32_bf16_k(const float* __restrict__ s, bf16* __restrict__ d) {
  size_t i = ((size_t)blockIdx.x * blockDim.x + threadIdx.x) * 4;
  v4f v = *(const v4f*)(s + i);
  v4bf r;
  r[0] = (bf16)v[0]; r[1] = (bf16)v[1]; r[2] = (bf16)v[2]; r[3] = (bf16)v[3];
  *(v4bf*)(d + i) = r;
}

// ---------------------------------------------------------------------------
// Up projection: h = silu(x @ w3^T) * (x @ w1^T), per-expert gathered rows.
// grid = (NEXP*MAXTT, FDIM/TN), block = 256 (8 waves, 4x2 wave tiling)
// X tile: per-lane async gather.  W1/W3 tiles: TDM (wave 0).
// ---------------------------------------------------------------------------
__global__ __launch_bounds__(256) void up_gemm_k(
    const bf16* __restrict__ xb,       // [NTOK][HDIM] bf16
    const bf16* __restrict__ w13b,     // [E][2F][H]  bf16
    const int*  __restrict__ bucket,
    const int*  __restrict__ counts,
    const int*  __restrict__ offsets,
    bf16*       __restrict__ hbuf) {   // [NPAIR][FDIM] bf16
  int e  = blockIdx.x / MAXTT;
  int tt = blockIdx.x % MAXTT;
  int ft = blockIdx.y;
  int cnt = counts[e];
  int rowBase = tt * TM;
  if (rowBase >= cnt) return;
  int valid = min(TM, cnt - rowBase);
  int slotBase = offsets[e] + rowBase;

  __shared__ bf16 sX[2][TM * KB];
  __shared__ bf16 sW1[2][TN * KB];
  __shared__ bf16 sW3[2][TN * KB];

  int tid  = threadIdx.x;
  int wave = tid >> 5, lane = tid & 31;
  int wm = wave >> 1, wn = wave & 1;        // 4x2 wave grid -> 32x32 per wave
  int lm = lane & 15, kh = (lane >> 4) << 3;

  int xr = tid >> 1, xc = (tid & 1) * 16;   // X tile: 128 rows x 32, 2 thr/row
  int tok = -1;
  if (xr < valid) tok = bucket[slotBase + xr] >> 1;  // pair -> token
  const bf16* xsrc   = xb + (size_t)tok * HDIM;
  const bf16* w1tile = w13b + ((size_t)e * 2 * FDIM + (size_t)ft * TN) * HDIM;
  const bf16* w3tile = w13b + ((size_t)e * 2 * FDIM + FDIM + (size_t)ft * TN) * HDIM;
#if !HAVE_TDM
  int wr = tid & 63, ws8 = (tid >> 6) * 8;
#endif

  // zero pad rows once; loads never touch them
  if (tok < 0) {
    v8bf z = v8bf{};
    *(v8bf*)&sX[0][xr * KB + xc] = z; *(v8bf*)&sX[0][xr * KB + xc + 8] = z;
    *(v8bf*)&sX[1][xr * KB + xc] = z; *(v8bf*)&sX[1][xr * KB + xc + 8] = z;
  }

  auto issue = [&](int bufI, int k0) {
    if (tok >= 0) {
      async_copy16(&sX[bufI][xr * KB + xc],     xsrc + k0 + xc);
      async_copy16(&sX[bufI][xr * KB + xc + 8], xsrc + k0 + xc + 8);
    }
#if HAVE_TDM
    if (wave == 0) {
      tdm_load_tile(w1tile + k0, &sW1[bufI][0], HDIM, TN, TN);
      tdm_load_tile(w3tile + k0, &sW3[bufI][0], HDIM, TN, TN);
    }
#else
    async_copy16(&sW1[bufI][wr * KB + ws8], w1tile + (size_t)wr * HDIM + k0 + ws8);
    async_copy16(&sW3[bufI][wr * KB + ws8], w3tile + (size_t)wr * HDIM + k0 + ws8);
#endif
  };

  v8f acc1[2][2] = {{v8f{}, v8f{}}, {v8f{}, v8f{}}};
  v8f acc3[2][2] = {{v8f{}, v8f{}}, {v8f{}, v8f{}}};

  auto compute = [&](int bufI) {
    v16bf a0 = load_frag(&sX[bufI][(wm * 32 + lm) * KB], kh);
    v16bf a1 = load_frag(&sX[bufI][(wm * 32 + 16 + lm) * KB], kh);
#pragma unroll
    for (int ni = 0; ni < 2; ++ni) {
      v16bf b1 = load_frag(&sW1[bufI][(wn * 32 + ni * 16 + lm) * KB], kh);
      v16bf b3 = load_frag(&sW3[bufI][(wn * 32 + ni * 16 + lm) * KB], kh);
      acc1[0][ni] = WMMA_BF16(acc1[0][ni], a0, b1);
      acc1[1][ni] = WMMA_BF16(acc1[1][ni], a1, b1);
      acc3[0][ni] = WMMA_BF16(acc3[0][ni], a0, b3);
      acc3[1][ni] = WMMA_BF16(acc3[1][ni], a1, b3);
    }
  };

  issue(0, 0);
  int buf = 0;
  for (int k0 = 0; k0 + KB < HDIM; k0 += KB, buf ^= 1) {
    issue(buf ^ 1, k0 + KB);     // prefetch next stage into the other buffer
#if HAVE_TDM
    WAIT_ASYNC(2);
    if (wave == 0) WAIT_TENSOR(2);
#else
    WAIT_ASYNC(4);
#endif
    __syncthreads();
    compute(buf);
    __syncthreads();
  }
  WAIT_ASYNC(0);
#if HAVE_TDM
  if (wave == 0) WAIT_TENSOR(0);
#endif
  __syncthreads();
  compute(buf);

  // epilogue: silu(acc3)*acc1 -> bf16 h (C/D layout: n = lane&15, m = r + kh)
#pragma unroll
  for (int mi = 0; mi < 2; ++mi) {
#pragma unroll
    for (int ni = 0; ni < 2; ++ni) {
#pragma unroll
      for (int r = 0; r < 8; ++r) {
        int lr = wm * 32 + mi * 16 + kh + r;
        if (lr < valid) {
          float a1v = acc1[mi][ni][r], a3v = acc3[mi][ni][r];
          float hv = a1v * (a3v / (1.0f + __expf(-a3v)));
          int col = ft * TN + wn * 32 + ni * 16 + lm;
          hbuf[(size_t)(slotBase + lr) * FDIM + col] = (bf16)hv;
        }
      }
    }
  }
}

// ---------------------------------------------------------------------------
// Down projection: y[t] += w_pair * (h @ w2^T)
// grid = (NEXP*MAXTT, HDIM/TN), block = 256
// H tile (contiguous rows) and W2 tile both via TDM; OOB rows zero-filled
// by tensor_dim1 = valid.
// ---------------------------------------------------------------------------
__global__ __launch_bounds__(256) void down_gemm_k(
    const bf16*  __restrict__ hbuf,    // [NPAIR][FDIM]
    const bf16*  __restrict__ w2b,     // [E][H][F] bf16
    const float* __restrict__ rw,      // [NPAIR]
    const int*   __restrict__ bucket,
    const int*   __restrict__ counts,
    const int*   __restrict__ offsets,
    float*       __restrict__ y) {     // [NTOK][HDIM]
  int e  = blockIdx.x / MAXTT;
  int tt = blockIdx.x % MAXTT;
  int jt = blockIdx.y;
  int cnt = counts[e];
  int rowBase = tt * TM;
  if (rowBase >= cnt) return;
  int valid = min(TM, cnt - rowBase);
  int slotBase = offsets[e] + rowBase;

  __shared__ int   sTok[TM];
  __shared__ float sWgt[TM];
  __shared__ bf16  sH[2][TM * KB];
  __shared__ bf16  sW[2][TN * KB];

  int tid  = threadIdx.x;
  int wave = tid >> 5, lane = tid & 31;
  int wm = wave >> 1, wn = wave & 1;
  int lm = lane & 15, kh = (lane >> 4) << 3;

  if (tid < TM) {
    int t2 = 0; float wv = 0.0f;
    if (tid < valid) {
      int p = bucket[slotBase + tid];
      t2 = p >> 1;
      wv = rw[p];
    }
    sTok[tid] = t2; sWgt[tid] = wv;
  }

  const bf16* htile = hbuf + (size_t)slotBase * FDIM;
  const bf16* wtile = w2b + ((size_t)e * HDIM + (size_t)jt * TN) * FDIM;

#if !HAVE_TDM
  int xr = tid >> 1, xc = (tid & 1) * 16;
  int wr = tid & 63, ws8 = (tid >> 6) * 8;
  if (xr >= valid) {
    v8bf z = v8bf{};
    *(v8bf*)&sH[0][xr * KB + xc] = z; *(v8bf*)&sH[0][xr * KB + xc + 8] = z;
    *(v8bf*)&sH[1][xr * KB + xc] = z; *(v8bf*)&sH[1][xr * KB + xc + 8] = z;
  }
#endif

  auto issue = [&](int bufI, int k0) {
#if HAVE_TDM
    if (wave == 0) {
      tdm_load_tile(htile + k0, &sH[bufI][0], FDIM, (unsigned)valid, TM);
      tdm_load_tile(wtile + k0, &sW[bufI][0], FDIM, TN, TN);
    }
#else
    if (xr < valid) {
      async_copy16(&sH[bufI][xr * KB + xc],     htile + (size_t)xr * FDIM + k0 + xc);
      async_copy16(&sH[bufI][xr * KB + xc + 8], htile + (size_t)xr * FDIM + k0 + xc + 8);
    }
    async_copy16(&sW[bufI][wr * KB + ws8], wtile + (size_t)wr * FDIM + k0 + ws8);
#endif
  };

  v8f acc[2][2] = {{v8f{}, v8f{}}, {v8f{}, v8f{}}};

  auto compute = [&](int bufI) {
    v16bf a0 = load_frag(&sH[bufI][(wm * 32 + lm) * KB], kh);
    v16bf a1 = load_frag(&sH[bufI][(wm * 32 + 16 + lm) * KB], kh);
#pragma unroll
    for (int ni = 0; ni < 2; ++ni) {
      v16bf b = load_frag(&sW[bufI][(wn * 32 + ni * 16 + lm) * KB], kh);
      acc[0][ni] = WMMA_BF16(acc[0][ni], a0, b);
      acc[1][ni] = WMMA_BF16(acc[1][ni], a1, b);
    }
  };

  issue(0, 0);
  int buf = 0;
  for (int k0 = 0; k0 + KB < FDIM; k0 += KB, buf ^= 1) {
    issue(buf ^ 1, k0 + KB);
#if HAVE_TDM
    if (wave == 0) WAIT_TENSOR(2);
#else
    WAIT_ASYNC(3);
#endif
    __syncthreads();
    compute(buf);
    __syncthreads();
  }
#if HAVE_TDM
  if (wave == 0) WAIT_TENSOR(0);
#else
  WAIT_ASYNC(0);
#endif
  __syncthreads();
  compute(buf);

#pragma unroll
  for (int mi = 0; mi < 2; ++mi) {
#pragma unroll
    for (int ni = 0; ni < 2; ++ni) {
#pragma unroll
      for (int r = 0; r < 8; ++r) {
        int lr = wm * 32 + mi * 16 + kh + r;
        if (lr < valid) {
          int   t2 = sTok[lr];
          float wv = sWgt[lr];
          int col = jt * TN + wn * 32 + ni * 16 + lm;
          atomicAdd(&y[(size_t)t2 * HDIM + col], wv * acc[mi][ni][r]);
        }
      }
    }
  }
}

// ---------------------------------------------------------------------------
// Workspace layout (bytes):
//   [0,32) counts[8]  [64,100) offsets[9]  [128,160) cursors[8]
//   [256,65792) bucket[16384]
//   XOFF=0x20000: x bf16 16MB | W13OFF: w13 bf16 64MB | W2OFF: w2 bf16 32MB
//   HOFF: h bf16 64MB    (total ~176.1 MB)
// ---------------------------------------------------------------------------
#define XOFF   ((size_t)0x20000)
#define W13OFF (XOFF + (size_t)16 * 1024 * 1024)
#define W2OFF  (W13OFF + (size_t)64 * 1024 * 1024)
#define HOFF   (W2OFF + (size_t)32 * 1024 * 1024)

extern "C" void kernel_launch(void* const* d_in, const int* in_sizes, int n_in,
                              void* d_out, int out_size, void* d_ws, size_t ws_size,
                              hipStream_t stream) {
  (void)in_sizes; (void)n_in; (void)ws_size;
  const float* x   = (const float*)d_in[0];
  const float* rw  = (const float*)d_in[1];
  const int*   sel = (const int*)d_in[2];
  const float* w13 = (const float*)d_in[3];
  const float* w2  = (const float*)d_in[4];
  float*       y   = (float*)d_out;

  char* ws = (char*)d_ws;
  int*  counts  = (int*)(ws + 0);
  int*  offsets = (int*)(ws + 64);
  int*  cursors = (int*)(ws + 128);
  int*  bucket  = (int*)(ws + 256);
  bf16* xb      = (bf16*)(ws + XOFF);
  bf16* w13b    = (bf16*)(ws + W13OFF);
  bf16* w2b     = (bf16*)(ws + W2OFF);
  bf16* hbuf    = (bf16*)(ws + HOFF);

  // routing
  zero_meta_k<<<1, 32, 0, stream>>>(counts, cursors);
  count_k<<<NPAIR / 256, 256, 0, stream>>>(sel, counts);
  scan_k<<<1, 32, 0, stream>>>(counts, offsets);
  scatter_k<<<NPAIR / 256, 256, 0, stream>>>(sel, offsets, cursors, bucket);

  // one-time fp32 -> bf16 conversions (x, w13, w2)
  cvt_f32_bf16_k<<<(NTOK * HDIM) / 1024, 256, 0, stream>>>(x, xb);
  cvt_f32_bf16_k<<<(NEXP * 2 * FDIM * HDIM) / 1024, 256, 0, stream>>>(w13, w13b);
  cvt_f32_bf16_k<<<(NEXP * HDIM * FDIM) / 1024, 256, 0, stream>>>(w2, w2b);

  // zero output (accumulated with atomics)
  (void)hipMemsetAsync(d_out, 0, (size_t)out_size * sizeof(float), stream);

  // grouped GEMMs
  dim3 gUp(NEXP * MAXTT, FDIM / TN);
  up_gemm_k<<<gUp, 256, 0, stream>>>(xb, w13b, bucket, counts, offsets, hbuf);

  dim3 gDn(NEXP * MAXTT, HDIM / TN);
  down_gemm_k<<<gDn, 256, 0, stream>>>(hbuf, w2b, rw, bucket, counts, offsets, y);
}
